// GATv2Model_53463752900678
// MI455X (gfx1250) — matrix operand order repaired and checked by reading, not compiled, not run on hardware
//
#include <hip/hip_runtime.h>
#include <hip/hip_bf16.h>
#include <math.h>

// ---------------------------------------------------------------------------
// GATv2 3-layer forward for MI455X (gfx1250, wave32, WMMA).
// Dense transforms via V_WMMA_F32_16X16X32_BF16 with 2x4 register blocking;
// edge softmax via L2-resident float atomics.
// ---------------------------------------------------------------------------

typedef __bf16  bf16_t;
typedef bf16_t  v16bf __attribute__((ext_vector_type(16)));
typedef bf16_t  v8bf  __attribute__((ext_vector_type(8)));
typedef float   v8f   __attribute__((ext_vector_type(8)));

constexpr int kN  = 20000;          // nodes
constexpr int kE  = 320000;         // edges (without self loops)
constexpr int kET = kE + kN;        // edges incl. self loops = 340000
constexpr int kIN = 512;            // input features
constexpr int kH  = 4;              // heads
constexpr int kC  = 64;             // channels per head
constexpr int kHC = kH * kC;        // 256

__device__ __forceinline__ unsigned short f32_to_bf16_rne(float f) {
  unsigned int u = __float_as_uint(f);
  u += 0x7FFFu + ((u >> 16) & 1u);
  return (unsigned short)(u >> 16);
}

// Sign-aware float atomic max using integer atomics (works for any sign mix,
// buffer pre-initialized to -inf).
__device__ __forceinline__ void atomic_max_f32(float* addr, float val) {
  if (val >= 0.0f) atomicMax((int*)addr, __float_as_int(val));
  else             atomicMin((unsigned int*)addr, __float_as_uint(val));
}

// ---------------------------------------------------------------------------
// Elementwise helpers
// ---------------------------------------------------------------------------
__global__ void fill_f32_k(float* __restrict__ p, float v, int n) {
  int i = blockIdx.x * blockDim.x + threadIdx.x;
  if (i < n) p[i] = v;
}

__global__ void f32_to_bf16_k(const float* __restrict__ src,
                              unsigned short* __restrict__ dst, int n) {
  int i = blockIdx.x * blockDim.x + threadIdx.x;
  if (i < n) dst[i] = f32_to_bf16_rne(src[i]);
}

// W [K, kHC] row-major f32  ->  WT [kHC, K] row-major bf16
__global__ void prep_w_k(const float* __restrict__ W,
                         unsigned short* __restrict__ WT, int K) {
  int idx = blockIdx.x * blockDim.x + threadIdx.x;
  if (idx >= K * kHC) return;
  int k = idx / kHC;
  int n = idx - k * kHC;
  WT[(size_t)n * K + k] = f32_to_bf16_rne(W[idx]);
}

// ---------------------------------------------------------------------------
// bf16 WMMA GEMM:  C[kN,kHC] (f32) = A[kN,K] (bf16) * BT[kHC,K]^T (bf16)
// One wave computes a 32x64 strip (2x4 tiles of 16x16), looping K in steps
// of 32: 2 A fragments + 4 B fragments -> 8 WMMAs per step.
// Fragment layouts (wave32): A/B use contiguous k = 16*half + s per lane half.
// ---------------------------------------------------------------------------
template <int K>
__global__ __launch_bounds__(256) void gemm_bf16_wmma(
    const unsigned short* __restrict__ A,
    const unsigned short* __restrict__ BT,
    float* __restrict__ Cg) {
  constexpr int TM = 2, TN = 4;                       // tiles per wave
  constexpr int nsn = (kHC / 16) / TN;                // 4 col strips
  constexpr int nsm = (kN / 16) / TM;                 // 625 row strips
  const int lane = threadIdx.x & 31;
  const int wave = threadIdx.x >> 5;
  const int strip = blockIdx.x * 8 + wave;            // 8 waves per block
  if (strip >= nsm * nsn) return;                     // wave-uniform predicate
  const int sm = strip / nsn;
  const int sn = strip - sm * nsn;

  const int half = lane >> 4;                         // lanes 0-15 / 16-31
  const int l15  = lane & 15;
  // A: row = strip row + (lane&15); lanes>=16 carry K [16..31], contiguous
  const unsigned short* Arow0 = A + (size_t)(sm * 32 + l15) * K + half * 16;
  const unsigned short* Arow1 = Arow0 + (size_t)16 * K;
  // B (from W^T): lane selects column; lanes>=16 carry K [16..31], contiguous
  const unsigned short* Brow0 = BT + (size_t)(sn * 64 + l15) * K + half * 16;

  v8f acc[TM][TN] = {};

  for (int k0 = 0; k0 < K; k0 += 32) {
    __builtin_prefetch(Arow0 + k0 + 256, 0, 0);        // global_prefetch_b8
    __builtin_prefetch(Arow1 + k0 + 256, 0, 0);
    union { v16bf v; v8bf h[2]; } a0, a1;
    a0.h[0] = *(const v8bf*)(Arow0 + k0);              // K = off .. off+7
    a0.h[1] = *(const v8bf*)(Arow0 + k0 + 8);          // K = off+8 .. off+15
    a1.h[0] = *(const v8bf*)(Arow1 + k0);
    a1.h[1] = *(const v8bf*)(Arow1 + k0 + 8);
#pragma unroll
    for (int j = 0; j < TN; ++j) {
      v16bf b = *(const v16bf*)(Brow0 + (size_t)(j * 16) * K + k0);
      acc[0][j] = __builtin_amdgcn_wmma_f32_16x16x32_bf16(
          false, a0.v, false, b, (short)0, acc[0][j], false, false);
      acc[1][j] = __builtin_amdgcn_wmma_f32_16x16x32_bf16(
          false, a1.v, false, b, (short)0, acc[1][j], false, false);
    }
  }

  // C layout: VGPR r -> (M = r, lanes 0-15) / (M = 8+r, lanes 16-31); N = l15
#pragma unroll
  for (int i = 0; i < TM; ++i) {
#pragma unroll
    for (int j = 0; j < TN; ++j) {
      float* Crow = Cg + (size_t)(sm * 32 + i * 16 + half * 8) * kHC
                       + sn * 64 + j * 16 + l15;
#pragma unroll
      for (int r = 0; r < 8; ++r) Crow[(size_t)r * kHC] = acc[i][j][r];
    }
  }
}

// ---------------------------------------------------------------------------
// Edge passes for layers 0/1 (H=4, C=64). One thread per (edge, head).
// Edge e >= kE is the self loop (e-kE, e-kE).
// ---------------------------------------------------------------------------
__device__ __forceinline__ void edge_sd(const int* __restrict__ ei, int e,
                                        int& s, int& d) {
  if (e < kE) { s = ei[e]; d = ei[kE + e]; }
  else        { s = e - kE; d = s; }
}

__global__ void edge_scores01_k(const float* __restrict__ XL,
                                const float* __restrict__ XR,
                                const float* __restrict__ att,
                                const int* __restrict__ ei,
                                float* __restrict__ esc,
                                float* __restrict__ emax) {
  int idx = blockIdx.x * blockDim.x + threadIdx.x;
  if (idx >= kET * kH) return;
  int e = idx >> 2, h = idx & 3;
  int s, d; edge_sd(ei, e, s, d);
  const float* xl = XL + (size_t)s * kHC + h * kC;
  const float* xr = XR + (size_t)d * kHC + h * kC;
  const float* at = att + h * kC;
  float acc = 0.f;
#pragma unroll 8
  for (int c = 0; c < kC; ++c) {
    float m = xl[c] + xr[c];
    m = (m > 0.f) ? m : 0.2f * m;               // LeakyReLU(0.2)
    acc = fmaf(m, at[c], acc);
  }
  esc[idx] = acc;
  atomic_max_f32(emax + (size_t)d * kH + h, acc);
}

__global__ void edge_exp01_k(const int* __restrict__ ei,
                             float* __restrict__ esc,
                             const float* __restrict__ emax,
                             float* __restrict__ denom) {
  int idx = blockIdx.x * blockDim.x + threadIdx.x;
  if (idx >= kET * kH) return;
  int e = idx >> 2, h = idx & 3;
  int s, d; edge_sd(ei, e, s, d);
  float ex = expf(esc[idx] - emax[(size_t)d * kH + h]);
  esc[idx] = ex;
  atomicAdd(denom + (size_t)d * kH + h, ex);
}

__global__ void edge_aggr01_k(const int* __restrict__ ei,
                              const float* __restrict__ esc,
                              const float* __restrict__ denom,
                              const float* __restrict__ XL,
                              float* __restrict__ hacc) {
  int idx = blockIdx.x * blockDim.x + threadIdx.x;
  if (idx >= kET * kH) return;
  int e = idx >> 2, h = idx & 3;
  int s, d; edge_sd(ei, e, s, d);
  float alpha = esc[idx] / (denom[(size_t)d * kH + h] + 1e-16f);
  const float* xl = XL + (size_t)s * kHC + h * kC;
  float* o = hacc + (size_t)d * kHC + h * kC;
#pragma unroll 8
  for (int c = 0; c < kC; ++c) atomicAdd(o + c, alpha * xl[c]);
}

// out = elu(bn(hacc + bias)); writes back f32 and bf16 copy for next GEMM
__global__ void bn_elu_k(float* __restrict__ hacc,
                         unsigned short* __restrict__ xb,
                         const float* __restrict__ b, const float* __restrict__ g,
                         const float* __restrict__ be, const float* __restrict__ m,
                         const float* __restrict__ v) {
  int i = blockIdx.x * blockDim.x + threadIdx.x;
  if (i >= kN * kHC) return;
  int j = i & (kHC - 1);
  float x = hacc[i] + b[j];
  x = (x - m[j]) * (g[j] * rsqrtf(v[j] + 1e-5f)) + be[j];
  x = (x > 0.f) ? x : expm1f(x);                // ELU
  hacc[i] = x;
  xb[i] = f32_to_bf16_rne(x);
}

// ---------------------------------------------------------------------------
// Layer 2: 256 -> 2, H=1, C=2 (scalar; negligible work)
// ---------------------------------------------------------------------------
__global__ void gemm2_k(const float* __restrict__ Hin,
                        const float* __restrict__ Wl, const float* __restrict__ Wr,
                        float* __restrict__ XL2, float* __restrict__ XR2) {
  int n = blockIdx.x * blockDim.x + threadIdx.x;
  if (n >= kN) return;
  const float* hr = Hin + (size_t)n * kHC;
  float l0 = 0.f, l1 = 0.f, r0 = 0.f, r1 = 0.f;
#pragma unroll 4
  for (int k = 0; k < kHC; ++k) {
    float hv = hr[k];
    l0 = fmaf(hv, Wl[2 * k], l0);     l1 = fmaf(hv, Wl[2 * k + 1], l1);
    r0 = fmaf(hv, Wr[2 * k], r0);     r1 = fmaf(hv, Wr[2 * k + 1], r1);
  }
  XL2[2 * n] = l0; XL2[2 * n + 1] = l1;
  XR2[2 * n] = r0; XR2[2 * n + 1] = r1;
}

__global__ void edge_scores2_k(const float* __restrict__ XL2,
                               const float* __restrict__ XR2,
                               const float* __restrict__ att2,
                               const int* __restrict__ ei,
                               float* __restrict__ esc2,
                               float* __restrict__ emax2) {
  int e = blockIdx.x * blockDim.x + threadIdx.x;
  if (e >= kET) return;
  int s, d; edge_sd(ei, e, s, d);
  float m0 = XL2[2 * s] + XR2[2 * d];         m0 = (m0 > 0.f) ? m0 : 0.2f * m0;
  float m1 = XL2[2 * s + 1] + XR2[2 * d + 1]; m1 = (m1 > 0.f) ? m1 : 0.2f * m1;
  float sc = fmaf(m0, att2[0], m1 * att2[1]);
  esc2[e] = sc;
  atomic_max_f32(emax2 + d, sc);
}

__global__ void edge_exp2_k(const int* __restrict__ ei, float* __restrict__ esc2,
                            const float* __restrict__ emax2,
                            float* __restrict__ den2) {
  int e = blockIdx.x * blockDim.x + threadIdx.x;
  if (e >= kET) return;
  int s, d; edge_sd(ei, e, s, d);
  float ex = expf(esc2[e] - emax2[d]);
  esc2[e] = ex;
  atomicAdd(den2 + d, ex);
}

__global__ void edge_aggr2_k(const int* __restrict__ ei,
                             const float* __restrict__ esc2,
                             const float* __restrict__ den2,
                             const float* __restrict__ XL2,
                             float* __restrict__ out2) {
  int e = blockIdx.x * blockDim.x + threadIdx.x;
  if (e >= kET) return;
  int s, d; edge_sd(ei, e, s, d);
  float a = esc2[e] / (den2[d] + 1e-16f);
  atomicAdd(out2 + 2 * d,     a * XL2[2 * s]);
  atomicAdd(out2 + 2 * d + 1, a * XL2[2 * s + 1]);
}

__global__ void final_k(const float* __restrict__ out2,
                        const float* __restrict__ b2, float* __restrict__ out) {
  int n = blockIdx.x * blockDim.x + threadIdx.x;
  if (n >= kN) return;
  float o0 = out2[2 * n] + b2[0];
  float o1 = out2[2 * n + 1] + b2[1];
  float mx = fmaxf(o0, o1);
  float l = logf(expf(o0 - mx) + expf(o1 - mx)) + mx;
  out[2 * n] = o0 - l;
  out[2 * n + 1] = o1 - l;
}

// ---------------------------------------------------------------------------
// Host launch
// ---------------------------------------------------------------------------
extern "C" void kernel_launch(void* const* d_in, const int* in_sizes, int n_in,
                              void* d_out, int out_size, void* d_ws, size_t ws_size,
                              hipStream_t stream) {
  (void)in_sizes; (void)n_in; (void)out_size; (void)ws_size;
  const float* x    = (const float*)d_in[0];
  const int*   ei   = (const int*)  d_in[1];
  const float* Wl0  = (const float*)d_in[2];
  const float* Wr0  = (const float*)d_in[3];
  const float* att0 = (const float*)d_in[4];
  const float* b0   = (const float*)d_in[5];
  const float* g0   = (const float*)d_in[6];
  const float* be0  = (const float*)d_in[7];
  const float* m0   = (const float*)d_in[8];
  const float* v0   = (const float*)d_in[9];
  const float* Wl1  = (const float*)d_in[10];
  const float* Wr1  = (const float*)d_in[11];
  const float* att1 = (const float*)d_in[12];
  const float* b1   = (const float*)d_in[13];
  const float* g1   = (const float*)d_in[14];
  const float* be1  = (const float*)d_in[15];
  const float* m1   = (const float*)d_in[16];
  const float* v1   = (const float*)d_in[17];
  const float* Wl2  = (const float*)d_in[18];
  const float* Wr2  = (const float*)d_in[19];
  const float* att2 = (const float*)d_in[20];
  const float* b2   = (const float*)d_in[21];
  float* out = (float*)d_out;

  // Workspace layout (bf16 regions 256B-aligned for 32B vector loads)
  char* ws = (char*)d_ws;
  unsigned short* XB   = (unsigned short*)(ws + 0);          // kN*kIN bf16 (act)
  unsigned short* WTa  = (unsigned short*)(ws + 20480000);   // Wl^T bf16
  unsigned short* WTb  = (unsigned short*)(ws + 20480000 + 262144);
  float* XL   = (float*)(ws + 21004288);                     // kN*kHC f32
  float* XR   = (float*)(ws + 41484288);
  float* HACC = (float*)(ws + 61964288);
  float* ESC  = (float*)(ws + 82444288);                     // kET*kH
  float* EMAX = (float*)(ws + 87884288);                     // kN*kH
  float* DEN  = (float*)(ws + 88204288);
  float* XL2  = (float*)(ws + 88524288);
  float* XR2  = (float*)(ws + 88684288);
  float* ESC2 = (float*)(ws + 88844288);
  float* EMAX2= (float*)(ws + 90204288);
  float* DEN2 = (float*)(ws + 90284288);
  float* OUT2 = (float*)(ws + 90364288);

  const int TPB = 256;
  auto cdiv = [](long long a, long long b) { return (int)((a + b - 1) / b); };
  // one wave per 32x64 strip: (1250/2 row strips) * (16/4 col strips) = 2500
  const int gemm_blocks = cdiv((kN / 32) * (kHC / 64), 8);
  const int egrid  = cdiv((long long)kET * kH, TPB);
  const int e2grid = cdiv(kET, TPB);
  const int nf     = cdiv((long long)kN * kHC, TPB);

  // ---------------- Layer 0 ----------------
  f32_to_bf16_k<<<cdiv((long long)kN * kIN, TPB), TPB, 0, stream>>>(x, XB, kN * kIN);
  prep_w_k<<<cdiv(kIN * kHC, TPB), TPB, 0, stream>>>(Wl0, WTa, kIN);
  prep_w_k<<<cdiv(kIN * kHC, TPB), TPB, 0, stream>>>(Wr0, WTb, kIN);
  gemm_bf16_wmma<kIN><<<gemm_blocks, 256, 0, stream>>>(XB, WTa, XL);
  gemm_bf16_wmma<kIN><<<gemm_blocks, 256, 0, stream>>>(XB, WTb, XR);
  fill_f32_k<<<cdiv(kN * kH, TPB), TPB, 0, stream>>>(EMAX, -INFINITY, kN * kH);
  fill_f32_k<<<cdiv(kN * kH, TPB), TPB, 0, stream>>>(DEN, 0.f, kN * kH);
  fill_f32_k<<<nf, TPB, 0, stream>>>(HACC, 0.f, kN * kHC);
  edge_scores01_k<<<egrid, TPB, 0, stream>>>(XL, XR, att0, ei, ESC, EMAX);
  edge_exp01_k<<<egrid, TPB, 0, stream>>>(ei, ESC, EMAX, DEN);
  edge_aggr01_k<<<egrid, TPB, 0, stream>>>(ei, ESC, DEN, XL, HACC);
  bn_elu_k<<<nf, TPB, 0, stream>>>(HACC, XB, b0, g0, be0, m0, v0);

  // ---------------- Layer 1 ----------------
  prep_w_k<<<cdiv(kHC * kHC, TPB), TPB, 0, stream>>>(Wl1, WTa, kHC);
  prep_w_k<<<cdiv(kHC * kHC, TPB), TPB, 0, stream>>>(Wr1, WTb, kHC);
  gemm_bf16_wmma<kHC><<<gemm_blocks, 256, 0, stream>>>(XB, WTa, XL);
  gemm_bf16_wmma<kHC><<<gemm_blocks, 256, 0, stream>>>(XB, WTb, XR);
  fill_f32_k<<<cdiv(kN * kH, TPB), TPB, 0, stream>>>(EMAX, -INFINITY, kN * kH);
  fill_f32_k<<<cdiv(kN * kH, TPB), TPB, 0, stream>>>(DEN, 0.f, kN * kH);
  fill_f32_k<<<nf, TPB, 0, stream>>>(HACC, 0.f, kN * kHC);
  edge_scores01_k<<<egrid, TPB, 0, stream>>>(XL, XR, att1, ei, ESC, EMAX);
  edge_exp01_k<<<egrid, TPB, 0, stream>>>(ei, ESC, EMAX, DEN);
  edge_aggr01_k<<<egrid, TPB, 0, stream>>>(ei, ESC, DEN, XL, HACC);
  bn_elu_k<<<nf, TPB, 0, stream>>>(HACC, XB, b1, g1, be1, m1, v1);

  // ---------------- Layer 2 (256 -> 2) ----------------
  gemm2_k<<<cdiv(kN, TPB), TPB, 0, stream>>>(HACC, Wl2, Wr2, XL2, XR2);
  fill_f32_k<<<cdiv(kN, TPB), TPB, 0, stream>>>(EMAX2, -INFINITY, kN);
  fill_f32_k<<<cdiv(kN, TPB), TPB, 0, stream>>>(DEN2, 0.f, kN);
  fill_f32_k<<<cdiv(2 * kN, TPB), TPB, 0, stream>>>(OUT2, 0.f, 2 * kN);
  edge_scores2_k<<<e2grid, TPB, 0, stream>>>(XL2, XR2, att2, ei, ESC2, EMAX2);
  edge_exp2_k<<<e2grid, TPB, 0, stream>>>(ei, ESC2, EMAX2, DEN2);
  edge_aggr2_k<<<e2grid, TPB, 0, stream>>>(ei, ESC2, DEN2, XL2, OUT2);
  final_k<<<cdiv(kN, TPB), TPB, 0, stream>>>(OUT2, b2, out);
}
